// FullyConnectedSE3_noR_55851754717695
// MI455X (gfx1250) — matrix operand order repaired and checked by reading, compiled, not run on hardware
//
#include <hip/hip_runtime.h>

#define L 512
#define DMSA 256
#define DPAIR 128
#define DRBF 64
#define C0 64
#define CE 64
#define CV 8
#define NH 4
#define NCHI 128
#define EK 192
#define KKP 160   // kin K padded (137 -> 160)
#define KNN 144   // kin N (64 | 64 | 16)

typedef __attribute__((ext_vector_type(16))) _Float16 v16h;
typedef __attribute__((ext_vector_type(8)))  _Float16 v8h;
typedef __attribute__((ext_vector_type(8)))  float    v8f;

enum ParamId {
  G_MSA, B_MSA, G_PAIR, B_PAIR, WN, BN, WN1, BN1, WN2, BN2, G_NODE, B_NODE,
  WE, BE, WE1, BE1, WE2, BE2, G_EDGE, B_EDGE, WVIN, WQ, WK, WV0, WG, WO0,
  G_LN, B_LN, WOV, WOUT0, BOUT0, WOUT1, NPARAM
};

struct Params { const float* p[NPARAM]; };

// ---------------- WMMA fragment helpers (wave32, 16x16x32 f16) ----------------
// A: row-major [M][K] in LDS, 16B-aligned rows; loads 2x ds_load_b128 per lane.
__device__ __forceinline__ v16h load_a_frag(const _Float16* A, int lda, int row0, int k0) {
  int lane = threadIdx.x & 31;
  const _Float16* Ar = A + (size_t)(row0 + (lane & 15)) * lda + (k0 + (lane >> 4) * 8);
  v8h lo = *(const v8h*)Ar;
  v8h hi = *(const v8h*)(Ar + 16);
  return __builtin_shufflevector(lo, hi, 0, 1, 2, 3, 4, 5, 6, 7, 8, 9, 10, 11, 12, 13, 14, 15);
}

// B stored TRANSPOSED in LDS: Bt[N][K]; a fragment reads K values 8h..8h+7 and 16+8h..16+8h+7
// for column (lane&15) -> 2x ds_load_b128 per lane instead of 16 scalar loads.
__device__ __forceinline__ v16h load_bt_frag(const _Float16* Bt, int ldb, int k0, int col0) {
  int lane = threadIdx.x & 31;
  const _Float16* Br = Bt + (size_t)(col0 + (lane & 15)) * ldb + (k0 + (lane >> 4) * 8);
  v8h lo = *(const v8h*)Br;
  v8h hi = *(const v8h*)(Br + 16);
  return __builtin_shufflevector(lo, hi, 0, 1, 2, 3, 4, 5, 6, 7, 8, 9, 10, 11, 12, 13, 14, 15);
}

// ---------------- K0: zero scratch ----------------
__global__ void zero_kernel(float* __restrict__ grad, unsigned* __restrict__ dmax_bits) {
  int t = blockIdx.x * blockDim.x + threadIdx.x;
  if (t < L * 9) grad[t] = 0.f;
  if (t == 0) *dmax_bits = 0u;
}

// ---------------- K1: node features s, s16, q ----------------
__global__ __launch_bounds__(64) void node_kernel(const float* __restrict__ msa, Params P,
                                                  float* __restrict__ s_out,
                                                  _Float16* __restrict__ s16_out,
                                                  float* __restrict__ q_out) {
  int i = blockIdx.x;
  int t = threadIdx.x;
  __shared__ float seq[DMSA];
  __shared__ float red[64];
  __shared__ float node[C0];
  __shared__ float h1[2 * C0];
  const float* row = msa + (size_t)i * DMSA;
  float part = 0.f;
  for (int k = t; k < DMSA; k += 64) { float x = row[k]; seq[k] = x; part += x; }
  red[t] = part; __syncthreads();
  for (int off = 32; off > 0; off >>= 1) { if (t < off) red[t] += red[t + off]; __syncthreads(); }
  float mean = red[0] * (1.f / DMSA);
  __syncthreads();
  part = 0.f;
  for (int k = t; k < DMSA; k += 64) { float d = seq[k] - mean; part += d * d; }
  red[t] = part; __syncthreads();
  for (int off = 32; off > 0; off >>= 1) { if (t < off) red[t] += red[t + off]; __syncthreads(); }
  float rstd = rsqrtf(red[0] * (1.f / DMSA) + 1e-5f);
  __syncthreads();
  for (int k = t; k < DMSA; k += 64) seq[k] = (seq[k] - mean) * rstd * P.p[G_MSA][k] + P.p[B_MSA][k];
  __syncthreads();
  float nv = P.p[BN][t];
  for (int k = 0; k < DMSA; ++k) nv += seq[k] * P.p[WN][k * C0 + t];
  node[t] = nv;
  __syncthreads();
  for (int c = t; c < 2 * C0; c += 64) {
    float hv = P.p[BN1][c];
    for (int k = 0; k < C0; ++k) hv += node[k] * P.p[WN1][k * 2 * C0 + c];
    h1[c] = fmaxf(hv, 0.f);
  }
  __syncthreads();
  float n2 = node[t] + P.p[BN2][t];
  for (int k = 0; k < 2 * C0; ++k) n2 += h1[k] * P.p[WN2][k * C0 + t];
  red[t] = n2; __syncthreads();
  for (int off = 32; off > 0; off >>= 1) { if (t < off) red[t] += red[t + off]; __syncthreads(); }
  float m2 = red[0] * (1.f / C0);
  __syncthreads();
  float dv = n2 - m2;
  red[t] = dv * dv; __syncthreads();
  for (int off = 32; off > 0; off >>= 1) { if (t < off) red[t] += red[t + off]; __syncthreads(); }
  float rstd2 = rsqrtf(red[0] * (1.f / C0) + 1e-5f);
  float sval = (n2 - m2) * rstd2 * P.p[G_NODE][t] + P.p[B_NODE][t];
  s_out[i * C0 + t] = sval;
  s16_out[i * C0 + t] = (_Float16)sval;
  __syncthreads();
  node[t] = sval;
  __syncthreads();
  float qv = 0.f;
  for (int k = 0; k < C0; ++k) qv += node[k] * P.p[WQ][k * C0 + t];
  q_out[i * C0 + t] = qv;
}

// ---------------- K2: analytic chiral-loss gradient ----------------
__global__ void chiral_kernel(const float* __restrict__ xyz, const float* __restrict__ chirals,
                              float* __restrict__ grad) {
  int n = blockIdx.x * blockDim.x + threadIdx.x;
  if (n >= NCHI) return;
  const float* ch = chirals + (size_t)n * 5;
  int ind[4];
#pragma unroll
  for (int k = 0; k < 4; ++k) {
    int iv = (int)ch[k];
    iv = iv < 0 ? 0 : iv;
    iv = iv > (L * 3 - 1) ? (L * 3 - 1) : iv;
    ind[k] = iv;
  }
  float tgt = ch[4];
  float p0[3], u[3][3], dn[3][3], inv[3];
#pragma unroll
  for (int d = 0; d < 3; ++d) p0[d] = xyz[(size_t)ind[0] * 3 + d];
#pragma unroll
  for (int k = 0; k < 3; ++k) {
    float nn = 1e-8f;
#pragma unroll
    for (int d = 0; d < 3; ++d) {
      u[k][d] = xyz[(size_t)ind[k + 1] * 3 + d] - p0[d];
      nn += u[k][d] * u[k][d];
    }
    inv[k] = 1.f / sqrtf(nn);
#pragma unroll
    for (int d = 0; d < 3; ++d) dn[k][d] = u[k][d] * inv[k];
  }
  float c23[3] = { dn[1][1] * dn[2][2] - dn[1][2] * dn[2][1],
                   dn[1][2] * dn[2][0] - dn[1][0] * dn[2][2],
                   dn[1][0] * dn[2][1] - dn[1][1] * dn[2][0] };
  float o = dn[0][0] * c23[0] + dn[0][1] * c23[1] + dn[0][2] * c23[2];
  float go = 2.f * (o - tgt);
  float gd[3][3];
#pragma unroll
  for (int d = 0; d < 3; ++d) gd[0][d] = go * c23[d];
  gd[1][0] = go * (dn[2][1] * dn[0][2] - dn[2][2] * dn[0][1]);
  gd[1][1] = go * (dn[2][2] * dn[0][0] - dn[2][0] * dn[0][2]);
  gd[1][2] = go * (dn[2][0] * dn[0][1] - dn[2][1] * dn[0][0]);
  gd[2][0] = go * (dn[0][1] * dn[1][2] - dn[0][2] * dn[1][1]);
  gd[2][1] = go * (dn[0][2] * dn[1][0] - dn[0][0] * dn[1][2]);
  gd[2][2] = go * (dn[0][0] * dn[1][1] - dn[0][1] * dn[1][0]);
  float g0[3] = {0.f, 0.f, 0.f};
#pragma unroll
  for (int k = 0; k < 3; ++k) {
    float dot = u[k][0] * gd[k][0] + u[k][1] * gd[k][1] + u[k][2] * gd[k][2];
    float sfac = dot * inv[k] * inv[k] * inv[k];
#pragma unroll
    for (int d = 0; d < 3; ++d) {
      float gu = inv[k] * gd[k][d] - sfac * u[k][d];
      atomicAdd(&grad[(size_t)ind[k + 1] * 3 + d], gu);
      g0[d] += gu;
    }
  }
#pragma unroll
  for (int d = 0; d < 3; ++d) atomicAdd(&grad[(size_t)ind[0] * 3 + d], -g0[d]);
}

// ---------------- K3: v = l1 * Wvin ----------------
__global__ void v_kernel(const float* __restrict__ grad, const float* __restrict__ Wvin,
                         float* __restrict__ v) {
  int t = blockIdx.x * blockDim.x + threadIdx.x;
  if (t >= L * CV * 3) return;
  int l = t / (CV * 3);
  int c = (t / 3) % CV;
  int dd = t % 3;
  v[t] = grad[(size_t)(l * 3 + 1) * 3 + dd] * Wvin[c];
}

// ---------------- K4: pairwise geometry d, rhat, max(d) ----------------
__global__ __launch_bounds__(256) void geom_kernel(const float* __restrict__ xyz,
                                                   float* __restrict__ darr,
                                                   float* __restrict__ rhat,
                                                   unsigned* __restrict__ dmax_bits) {
  int i = blockIdx.x;
  int t = threadIdx.x;
  __shared__ float ci[3];
  __shared__ float rmax[256];
  if (t < 3) ci[t] = xyz[(size_t)(i * 3 + 1) * 3 + t];
  __syncthreads();
  float lm = 0.f;
  for (int j = t; j < L; j += 256) {
    const float* cj = xyz + (size_t)(j * 3 + 1) * 3;
    float rx = cj[0] - ci[0], ry = cj[1] - ci[1], rz = cj[2] - ci[2];
    float d = sqrtf(rx * rx + ry * ry + rz * rz + 1e-8f);
    size_t po = (size_t)i * L + j;
    darr[po] = d;
    float inv = 1.f / d;
    rhat[po * 3 + 0] = rx * inv;
    rhat[po * 3 + 1] = ry * inv;
    rhat[po * 3 + 2] = rz * inv;
    lm = fmaxf(lm, d);
  }
  rmax[t] = lm; __syncthreads();
  for (int off = 128; off > 0; off >>= 1) { if (t < off) rmax[t] = fmaxf(rmax[t], rmax[t + off]); __syncthreads(); }
  if (t == 0) atomicMax(dmax_bits, __float_as_uint(rmax[0]));
}

// ---------------- K5: edge embed (WMMA GEMM chain + LN), 32 pairs per WG ----------------
__global__ __launch_bounds__(256) void edge_kernel(const float* __restrict__ pair,
                                                   const float* __restrict__ darr,
                                                   Params P,
                                                   _Float16* __restrict__ edge16) {
  int i = blockIdx.x >> 4;
  int j0 = (blockIdx.x & 15) << 5;
  int t = threadIdx.x;
  int wave = t >> 5;
  int lane = t & 31;
  __shared__ __align__(16) _Float16 sA[32][EK];     // features / reused A (rows 384B)
  __shared__ __align__(16) _Float16 sBt[64 * EK];   // transposed weights Bt[N][K]
  __shared__ __align__(16) float sE0[32][64];       // f32 edge0 residual / final pre-LN
  __shared__ __align__(16) _Float16 sH[32][128];    // hidden (rows 256B)
  __shared__ float sNei[32];
  __shared__ float sP1[32][8];
  __shared__ float sP2[32][8];
  int m = t >> 3, sub = t & 7;
  int j = j0 + m;
  const float* prow = pair + ((size_t)i * L + j) * DPAIR;
  float vals[16];
  float lsum = 0.f;
#pragma unroll
  for (int k = 0; k < 16; ++k) { float x = prow[sub * 16 + k]; vals[k] = x; lsum += x; }
  sP1[m][sub] = lsum;
  __syncthreads();
  float mean = 0.f;
#pragma unroll
  for (int k = 0; k < 8; ++k) mean += sP1[m][k];
  mean *= (1.f / DPAIR);
  float ls2 = 0.f;
#pragma unroll
  for (int k = 0; k < 16; ++k) { float d = vals[k] - mean; ls2 += d * d; }
  sP2[m][sub] = ls2;
  __syncthreads();
  float var = 0.f;
#pragma unroll
  for (int k = 0; k < 8; ++k) var += sP2[m][k];
  float rstd = rsqrtf(var * (1.f / DPAIR) + 1e-5f);
#pragma unroll
  for (int k = 0; k < 16; ++k) {
    int c = sub * 16 + k;
    sA[m][c] = (_Float16)((vals[k] - mean) * rstd * P.p[G_PAIR][c] + P.p[B_PAIR][c]);
  }
  float dv = darr[(size_t)i * L + j];
#pragma unroll
  for (int k = 0; k < 8; ++k) {
    int c = sub * 8 + k;
    float mu = 2.f + 20.f * (float)c / 63.f;
    float z = (dv - mu) * (1.f / 0.3125f);
    sA[m][128 + c] = (_Float16)expf(-z * z);
  }
  if (sub == 0) {
    float off = (float)(j - i);
    float sg = (off > 0.f) ? 1.f : ((off < 0.f) ? -1.f : 0.f);
    sNei[m] = sg * log1pf(fabsf(off));
  }
  // stage We transposed: Bt[n][k], n<64, k<192
  for (int idx = t; idx < 64 * EK; idx += 256) {
    int n = idx / EK, k = idx - n * EK;
    sBt[idx] = (_Float16)P.p[WE][k * 64 + n];
  }
  __syncthreads();
  // GEMM1: edge0 = featE(192) @ We + be + neigh * We[192,:]
  {
    int mt = wave >> 2, nt = wave & 3;
    int row0 = mt * 16;
    int col = lane & 15;
    int colg = nt * 16 + col;
    int rbase = (lane >> 4) * 8;
    const float* we192 = P.p[WE] + 192 * 64;
    v8f c;
#pragma unroll
    for (int r = 0; r < 8; ++r) c[r] = P.p[BE][colg] + sNei[row0 + rbase + r] * we192[colg];
#pragma unroll
    for (int kb = 0; kb < 6; ++kb) {
      v16h a = load_a_frag(&sA[0][0], EK, row0, kb * 32);
      v16h b = load_bt_frag(sBt, EK, kb * 32, nt * 16);
      c = __builtin_amdgcn_wmma_f32_16x16x32_f16(false, a, false, b, (short)0, c, false, false);
    }
#pragma unroll
    for (int r = 0; r < 8; ++r) sE0[row0 + rbase + r][colg] = c[r];
  }
  __syncthreads();
  for (int idx = t; idx < 32 * 64; idx += 256) {
    int mm = idx >> 6, cc = idx & 63;
    sA[mm][cc] = (_Float16)sE0[mm][cc];
  }
  // stage We1 transposed: Bt[n][k], n<128, k<64
  for (int idx = t; idx < 128 * 64; idx += 256) {
    int n = idx >> 6, k = idx & 63;
    sBt[idx] = (_Float16)P.p[WE1][k * 128 + n];
  }
  __syncthreads();
  // GEMM2: h = relu(edge0 @ We1 + be1)
  for (int job = wave; job < 16; job += 8) {
    int mt = job >> 3, nt = job & 7;
    int row0 = mt * 16;
    int col = lane & 15;
    int colg = nt * 16 + col;
    int rbase = (lane >> 4) * 8;
    v8f c;
#pragma unroll
    for (int r = 0; r < 8; ++r) c[r] = P.p[BE1][colg];
#pragma unroll
    for (int kb = 0; kb < 2; ++kb) {
      v16h a = load_a_frag(&sA[0][0], EK, row0, kb * 32);
      v16h b = load_bt_frag(sBt, 64, kb * 32, nt * 16);
      c = __builtin_amdgcn_wmma_f32_16x16x32_f16(false, a, false, b, (short)0, c, false, false);
    }
#pragma unroll
    for (int r = 0; r < 8; ++r) sH[row0 + rbase + r][colg] = (_Float16)fmaxf((float)c[r], 0.f);
  }
  __syncthreads();
  // stage We2 transposed: Bt[n][k], n<64, k<128
  for (int idx = t; idx < 64 * 128; idx += 256) {
    int n = idx >> 7, k = idx & 127;
    sBt[idx] = (_Float16)P.p[WE2][k * 64 + n];
  }
  __syncthreads();
  // GEMM3: edge = edge0 + h @ We2 + be2
  {
    int mt = wave >> 2, nt = wave & 3;
    int row0 = mt * 16;
    int col = lane & 15;
    int colg = nt * 16 + col;
    int rbase = (lane >> 4) * 8;
    v8f c;
#pragma unroll
    for (int r = 0; r < 8; ++r) c[r] = P.p[BE2][colg] + sE0[row0 + rbase + r][colg];
#pragma unroll
    for (int kb = 0; kb < 4; ++kb) {
      v16h a = load_a_frag(&sH[0][0], 128, row0, kb * 32);
      v16h b = load_bt_frag(sBt, 128, kb * 32, nt * 16);
      c = __builtin_amdgcn_wmma_f32_16x16x32_f16(false, a, false, b, (short)0, c, false, false);
    }
#pragma unroll
    for (int r = 0; r < 8; ++r) sE0[row0 + rbase + r][colg] = c[r];
  }
  __syncthreads();
  // LayerNorm per row + store f16
  float esum = 0.f;
  float ev[8];
#pragma unroll
  for (int k = 0; k < 8; ++k) { ev[k] = sE0[m][sub * 8 + k]; esum += ev[k]; }
  sP1[m][sub] = esum;
  __syncthreads();
  float em = 0.f;
#pragma unroll
  for (int k = 0; k < 8; ++k) em += sP1[m][k];
  em *= (1.f / CE);
  float es2 = 0.f;
#pragma unroll
  for (int k = 0; k < 8; ++k) { float d = ev[k] - em; es2 += d * d; }
  sP2[m][sub] = es2;
  __syncthreads();
  float evar = 0.f;
#pragma unroll
  for (int k = 0; k < 8; ++k) evar += sP2[m][k];
  float erstd = rsqrtf(evar * (1.f / CE) + 1e-5f);
  size_t base = ((size_t)i * L + j) * 64;
#pragma unroll
  for (int k = 0; k < 8; ++k) {
    int c = sub * 8 + k;
    edge16[base + c] = (_Float16)((ev[k] - em) * erstd * P.p[G_EDGE][c] + P.p[B_EDGE][c]);
  }
}

// ---------------- K6: kin GEMM -> logits, val0, g (WMMA, fused Wk|Wv0|Wg) ----------------
__global__ __launch_bounds__(256) void kin_kernel(Params P,
    const _Float16* __restrict__ s16, const _Float16* __restrict__ edge16,
    const float* __restrict__ vbuf, const float* __restrict__ rhat,
    const float* __restrict__ darr, const unsigned* __restrict__ dmax_bits,
    const float* __restrict__ qbuf, float* __restrict__ logits,
    _Float16* __restrict__ val016, _Float16* __restrict__ g16) {
  int i = blockIdx.x >> 4;
  int j0 = (blockIdx.x & 15) << 5;
  int t = threadIdx.x;
  int wave = t >> 5, lane = t & 31;
  __shared__ __align__(16) _Float16 sA[32][KKP];     // rows 320B
  __shared__ __align__(16) _Float16 sBt[KNN * KKP];  // Bt[n][k], n<144, k<160
  __shared__ float sQ[64];
  if (t < 64) sQ[t] = qbuf[i * 64 + t];
  // stage fused weights transposed + zero K padding
  for (int idx = t; idx < KNN * KKP; idx += 256) {
    int n = idx / KKP, k = idx - n * KKP;
    float w = 0.f;
    if (k < 137) {
      if (n < 64) w = P.p[WK][k * 64 + n];
      else if (n < 128) w = P.p[WV0][k * 64 + (n - 64)];
      else w = P.p[WG][k * 16 + (n - 128)];
    }
    sBt[idx] = (_Float16)w;
  }
  int m = t >> 3, sub = t & 7;
  int j = j0 + m;
  size_t po = (size_t)i * L + j;
#pragma unroll
  for (int k = 0; k < 8; ++k) sA[m][sub * 8 + k] = s16[j * 64 + sub * 8 + k];
#pragma unroll
  for (int k = 0; k < 8; ++k) sA[m][64 + sub * 8 + k] = edge16[po * 64 + sub * 8 + k];
  {
    float r0 = rhat[po * 3 + 0], r1 = rhat[po * 3 + 1], r2 = rhat[po * 3 + 2];
    const float* vj = vbuf + j * 24 + sub * 3;
    sA[m][128 + sub] = (_Float16)(vj[0] * r0 + vj[1] * r1 + vj[2] * r2);
  }
  if (sub == 0) {
    float dmax = __uint_as_float(*dmax_bits);
    float x = darr[po] / (dmax + 1e-6f);
    x = fminf(fmaxf(x, 0.f), 1.f);
    sA[m][136] = (_Float16)asinf(x);
#pragma unroll
    for (int k = 137; k < KKP; ++k) sA[m][k] = (_Float16)0.f;
  }
  __syncthreads();
  for (int job = wave; job < 18; job += 8) {
    int mt = job / 9, nt = job - mt * 9;
    int row0 = mt * 16;
    int col = lane & 15;
    int rbase = (lane >> 4) * 8;
    v8f c;
#pragma unroll
    for (int r = 0; r < 8; ++r) c[r] = 0.f;
#pragma unroll
    for (int kb = 0; kb < 5; ++kb) {
      v16h a = load_a_frag(&sA[0][0], KKP, row0, kb * 32);
      v16h b = load_bt_frag(sBt, KKP, kb * 32, nt * 16);
      c = __builtin_amdgcn_wmma_f32_16x16x32_f16(false, a, false, b, (short)0, c, false, false);
    }
    if (nt < 4) {
      // head nt: logits[nt,i,j] = (q . k)/4 via half-wave shuffle reduce
      float qv = sQ[nt * 16 + col];
#pragma unroll
      for (int r = 0; r < 8; ++r) {
        float val = c[r] * qv;
        val += __shfl_xor(val, 1);
        val += __shfl_xor(val, 2);
        val += __shfl_xor(val, 4);
        val += __shfl_xor(val, 8);
        if (col == 0) {
          int jr = j0 + row0 + rbase + r;
          float lg = val * 0.25f;
          if (jr == i) lg = -1e9f;
          logits[((size_t)nt * L + i) * L + jr] = lg;
        }
      }
    } else if (nt < 8) {
      int cc = (nt - 4) * 16 + col;
#pragma unroll
      for (int r = 0; r < 8; ++r) {
        int jr = j0 + row0 + rbase + r;
        val016[((size_t)i * L + jr) * 64 + cc] = (_Float16)c[r];
      }
    } else {
#pragma unroll
      for (int r = 0; r < 8; ++r) {
        int jr = j0 + row0 + rbase + r;
        g16[((size_t)i * L + jr) * 16 + col] = (_Float16)c[r];
      }
    }
  }
}

// ---------------- K7: softmax over j ----------------
__global__ __launch_bounds__(128) void softmax_kernel(float* __restrict__ logits) {
  size_t row = blockIdx.x;
  float* ptr = logits + row * L;
  int t = threadIdx.x;
  __shared__ float red[128];
  float v0 = ptr[t], v1 = ptr[t + 128], v2 = ptr[t + 256], v3 = ptr[t + 384];
  float mx = fmaxf(fmaxf(v0, v1), fmaxf(v2, v3));
  red[t] = mx; __syncthreads();
  for (int off = 64; off > 0; off >>= 1) { if (t < off) red[t] = fmaxf(red[t], red[t + off]); __syncthreads(); }
  mx = red[0]; __syncthreads();
  float e0 = expf(v0 - mx), e1 = expf(v1 - mx), e2 = expf(v2 - mx), e3 = expf(v3 - mx);
  red[t] = e0 + e1 + e2 + e3; __syncthreads();
  for (int off = 64; off > 0; off >>= 1) { if (t < off) red[t] += red[t + off]; __syncthreads(); }
  float inv = 1.f / red[0];
  ptr[t] = e0 * inv; ptr[t + 128] = e1 * inv; ptr[t + 256] = e2 * inv; ptr[t + 384] = e3 * inv;
}

// ---------------- K8: m0 (attn x val0) and m1 (gated vector message) ----------------
__global__ __launch_bounds__(256) void m0m1_kernel(const float* __restrict__ attn,
    const _Float16* __restrict__ val016, const _Float16* __restrict__ g16,
    const float* __restrict__ vbuf, const float* __restrict__ rhat,
    float* __restrict__ m0, float* __restrict__ m1) {
  int i = blockIdx.x;
  int t = threadIdx.x;
  __shared__ float sa[L];
  __shared__ float red0[4][64];
  __shared__ float red1[24][4];
  for (int j = t; j < L; j += 256) {
    float aa = 0.f;
#pragma unroll
    for (int h = 0; h < NH; ++h) aa += attn[((size_t)h * L + i) * L + j];
    sa[j] = aa * 0.25f;
  }
  __syncthreads();
  {
    int hd = t & 63, grp = t >> 6;
    const float* at = attn + ((size_t)(hd >> 4) * L + i) * L;
    const _Float16* vp = val016 + ((size_t)i * L) * 64 + hd;
    float acc = 0.f;
    int jb = grp * 128;
    for (int j = jb; j < jb + 128; ++j) acc += at[j] * (float)vp[(size_t)j * 64];
    red0[grp][hd] = acc;
  }
  __syncthreads();
  if (t < 64) m0[i * 64 + t] = red0[0][t] + red0[1][t] + red0[2][t] + red0[3][t];
  if (t < 96) {
    int od = t >> 2, grp = t & 3;
    int c = od / 3, dd = od - c * 3;
    float acc = 0.f;
    int jb = grp * 128;
    for (int j = jb; j < jb + 128; ++j) {
      size_t po = (size_t)i * L + j;
      float gv = (float)g16[po * 16 + c];
      float gs = (float)g16[po * 16 + 8 + c];
      acc += sa[j] * (gv * vbuf[j * 24 + c * 3 + dd] + gs * rhat[po * 3 + dd]);
    }
    red1[od][grp] = acc;
  }
  __syncthreads();
  if (t < 24) m1[i * 24 + t] = red1[t][0] + red1[t][1] + red1[t][2] + red1[t][3];
}

// ---------------- K9: output heads ----------------
__global__ __launch_bounds__(64) void final_kernel(Params P,
    const float* __restrict__ sbuf, const float* __restrict__ m0,
    const float* __restrict__ m1, const float* __restrict__ vbuf,
    const float* __restrict__ state, const float* __restrict__ xyz,
    float* __restrict__ out) {
  int i = blockIdx.x;
  int t = threadIdx.x;
  __shared__ float sm0[64];
  __shared__ float red[64];
  __shared__ float sln[64];
  __shared__ float vfin[24];
  __shared__ float T3[3];
  sm0[t] = m0[i * 64 + t];
  __syncthreads();
  float x = sbuf[i * 64 + t];
  for (int k = 0; k < 64; ++k) x += sm0[k] * P.p[WO0][k * 64 + t];
  red[t] = x; __syncthreads();
  for (int off = 32; off > 0; off >>= 1) { if (t < off) red[t] += red[t + off]; __syncthreads(); }
  float mean = red[0] * (1.f / 64.f);
  __syncthreads();
  float d = x - mean;
  red[t] = d * d; __syncthreads();
  for (int off = 32; off > 0; off >>= 1) { if (t < off) red[t] += red[t + off]; __syncthreads(); }
  float rstd = rsqrtf(red[0] * (1.f / 64.f) + 1e-5f);
  sln[t] = (x - mean) * rstd * P.p[G_LN][t] + P.p[B_LN][t];
  __syncthreads();
  float o0 = P.p[BOUT0][t];
  for (int k = 0; k < 64; ++k) o0 += sln[k] * P.p[WOUT0][k * 64 + t];
  out[i * 64 + t] = state[i * 64 + t] + o0;
  if (t < 24) {
    int e = t / 3, dd = t - e * 3;
    float add = 0.f;
    for (int cc = 0; cc < 8; ++cc) add += m1[i * 24 + cc * 3 + dd] * P.p[WOV][cc * 8 + e];
    vfin[t] = vbuf[i * 24 + t] + add;
  }
  __syncthreads();
  if (t < 3) {
    float o1 = 0.f;
    for (int cc = 0; cc < 8; ++cc) o1 += vfin[cc * 3 + t] * P.p[WOUT1][cc];
    T3[t] = o1 * 0.1f;
  }
  __syncthreads();
  if (t < 9) {
    float val = xyz[(size_t)i * 9 + t];
    if (t >= 3 && t < 6) val += T3[t - 3];
    out[(size_t)L * 64 + i * 9 + t] = val;
  }
}

// ---------------- host launcher ----------------
extern "C" void kernel_launch(void* const* d_in, const int* in_sizes, int n_in,
                              void* d_out, int out_size, void* d_ws, size_t ws_size,
                              hipStream_t stream) {
  (void)out_size; (void)ws_size;
  const float* msa     = (const float*)d_in[0];
  const float* pair    = (const float*)d_in[1];
  const float* state   = (const float*)d_in[2];
  const float* xyz     = (const float*)d_in[3];
  const float* chirals = (const float*)d_in[5];

  static const int psz[NPARAM] = {256,256,128,128,16384,64,8192,128,8192,64,64,64,
                                  12352,64,8192,128,8192,64,64,64,8,4096,8768,8768,
                                  2192,4096,64,64,64,4096,64,8};
  Params P;
  if (n_in >= 38) {
    if (in_sizes[6] == 256) {
      // params expanded as leaves in dict insertion order
      for (int k = 0; k < NPARAM; ++k) P.p[k] = (const float*)d_in[6 + k];
    } else {
      // params expanded as leaves in alphabetically-sorted key order
      static const int sortmap[NPARAM] = {
        WE, WE1, WE2, WG, WK, WN, WN1, WN2, WO0, WOUT0, WOUT1, WOV, WQ, WV0, WVIN,
        B_EDGE, B_LN, B_MSA, B_NODE, B_PAIR, BE, BE1, BE2, BN, BN1, BN2, BOUT0,
        G_EDGE, G_LN, G_MSA, G_NODE, G_PAIR };
      for (int k = 0; k < NPARAM; ++k) P.p[sortmap[k]] = (const float*)d_in[6 + k];
    }
  } else {
    // params packed as one flat blob in insertion order
    const float* base = (const float*)d_in[6];
    size_t off = 0;
    for (int k = 0; k < NPARAM; ++k) { P.p[k] = base + off; off += (size_t)psz[k]; }
  }

  // workspace carve-out (256B aligned regions)
  char* w = (char*)d_ws;
  auto alloc = [&](size_t bytes) -> char* {
    char* r = w;
    w += (bytes + 255) & ~(size_t)255;
    return r;
  };
  float*     s      = (float*)alloc((size_t)L * 64 * 4);
  float*     q      = (float*)alloc((size_t)L * 64 * 4);
  float*     v      = (float*)alloc((size_t)L * 24 * 4);
  float*     grad   = (float*)alloc((size_t)L * 9 * 4);
  float*     darr   = (float*)alloc((size_t)L * L * 4);
  float*     rhat   = (float*)alloc((size_t)L * L * 3 * 4);
  float*     m0     = (float*)alloc((size_t)L * 64 * 4);
  float*     m1     = (float*)alloc((size_t)L * 24 * 4);
  float*     logits = (float*)alloc((size_t)NH * L * L * 4);
  unsigned*  dmaxb  = (unsigned*)alloc(256);
  _Float16*  s16    = (_Float16*)alloc((size_t)L * 64 * 2);
  _Float16*  edge16 = (_Float16*)alloc((size_t)L * L * 64 * 2);
  _Float16*  val016 = (_Float16*)alloc((size_t)L * L * 64 * 2);
  _Float16*  g16    = (_Float16*)alloc((size_t)L * L * 16 * 2);

  zero_kernel<<<(L * 9 + 255) / 256, 256, 0, stream>>>(grad, dmaxb);
  node_kernel<<<L, 64, 0, stream>>>(msa, P, s, s16, q);
  chiral_kernel<<<1, 128, 0, stream>>>(xyz, chirals, grad);
  v_kernel<<<(L * CV * 3 + 255) / 256, 256, 0, stream>>>(grad, P.p[WVIN], v);
  geom_kernel<<<L, 256, 0, stream>>>(xyz, darr, rhat, dmaxb);
  edge_kernel<<<L * 16, 256, 0, stream>>>(pair, darr, P, edge16);
  kin_kernel<<<L * 16, 256, 0, stream>>>(P, s16, edge16, v, rhat, darr, dmaxb, q,
                                         logits, val016, g16);
  softmax_kernel<<<NH * L, 128, 0, stream>>>(logits);
  m0m1_kernel<<<L, 256, 0, stream>>>(logits, val016, g16, v, rhat, m0, m1);
  final_kernel<<<L, 64, 0, stream>>>(P, s, m0, m1, v, state, xyz, (float*)d_out);
}
